// FasterRCNN_46557445489294
// MI455X (gfx1250) — matrix operand order repaired and checked by reading, not compiled, not run on hardware
//
#include <hip/hip_runtime.h>
#include <hip/hip_bf16.h>

// ---------------------------------------------------------------------------
// Types for CDNA5 WMMA (wave32) + TDM descriptor groups
// ---------------------------------------------------------------------------
typedef __attribute__((ext_vector_type(16))) __bf16 v16bf;
typedef __attribute__((ext_vector_type(8)))  float  v8f;
typedef __attribute__((ext_vector_type(4)))  unsigned u32x4;
typedef __attribute__((ext_vector_type(8)))  int      i32x8;
typedef __attribute__((ext_vector_type(4)))  int      i32x4;
typedef unsigned long long ull;

#if defined(__has_builtin)
# if __has_builtin(__builtin_amdgcn_tensor_load_to_lds)
#  define HAVE_TDM 1
# endif
#endif
#ifndef HAVE_TDM
# define HAVE_TDM 0
#endif

__device__ __forceinline__ __bf16 f2bf(float f) {
    unsigned u = __builtin_bit_cast(unsigned, f);
    unsigned short h = (unsigned short)(u >> 16);   // truncate-round to bf16
    return __builtin_bit_cast(__bf16, h);
}

__device__ __forceinline__ unsigned sortable_u32(float f) {
    unsigned u = __builtin_bit_cast(unsigned, f);
    return u ^ ((u & 0x80000000u) ? 0xFFFFFFFFu : 0x80000000u);
}

// Fragment loaders per CDNA5 ISA 7.12.2 VGPR layouts (wave32).
//   A (16x32 bf16): lane L -> row L&15; K chunks at (L>>4)*8 and +16.
//   B (32x16 bf16): lane L -> col L&15; K chunks at (L>>4)*8 and +16.
__device__ __forceinline__ v16bf load_a_frag(const __bf16* At, int lda, int lane) {
    const int mr = lane & 15, ak = (lane >> 4) * 8;
    v16bf a;
#pragma unroll
    for (int t = 0; t < 8; ++t) {
        a[t]     = At[mr * lda + ak + t];
        a[t + 8] = At[mr * lda + ak + 16 + t];
    }
    return a;
}
__device__ __forceinline__ v16bf load_b_frag(const __bf16* Bt, int ldb, int lane) {
    const int nc = lane & 15, bk = (lane >> 4) * 8;
    v16bf b;
#pragma unroll
    for (int t = 0; t < 8; ++t) {
        b[t]     = Bt[nc * ldb + bk + t];
        b[t + 8] = Bt[nc * ldb + bk + 16 + t];
    }
    return b;
}

// ---------------------------------------------------------------------------
// Constants
// ---------------------------------------------------------------------------
#define FH 50
#define FW 50
#define NPOS (FH * FW)
#define CIN 1024
#define CMID 512
#define NANCH (NPOS * 9)
#define NSORT 32768
#define NPRE 3000
#define NPOST 300
#define NMSW ((NPRE + 31) / 32)
#define POOLK (CIN * 49)
#define NFC 4096
#define OUT_REG 0
#define OUT_CLS 25200
#define OUT_LOC 31500
#define OUT_SCORE 121500
#define KSPLIT 8
#define MT_PER_WAVE 10          // 2 waves in M x 10 tiles x 16 rows = 320 padded rows

// ---------------------------------------------------------------------------
// Pre-pass: NCHW f32 -> NHWC bf16 features; W_conv -> [tap][oc][c] bf16
// ---------------------------------------------------------------------------
__global__ void featT_kernel(const float* __restrict__ feat, __bf16* __restrict__ featT) {
    int t = blockIdx.x * blockDim.x + threadIdx.x;
    if (t >= NPOS * CIN) return;
    int c = t % CIN, p = t / CIN;
    featT[t] = f2bf(feat[c * NPOS + p]);
}

__global__ void wconv_reorder_kernel(const float* __restrict__ W, __bf16* __restrict__ Wt) {
    int t = blockIdx.x * blockDim.x + threadIdx.x;
    if (t >= 9 * CMID * CIN) return;
    int c = t % CIN, r = t / CIN;
    int oc = r % CMID, tap = r / CMID;
    Wt[t] = f2bf(W[(oc * CIN + c) * 9 + tap]);
}

// ---------------------------------------------------------------------------
// Kernel 1: 3x3 conv (1024->512) + ReLU, implicit GEMM, tap-major K.
// 8 waves = 2(M)x4(N): 32 positions x 64 oc per workgroup.
// ---------------------------------------------------------------------------
__global__ void conv3x3_wmma_kernel(const __bf16* __restrict__ featT,
                                    const __bf16* __restrict__ Wt,
                                    const float* __restrict__ bias,
                                    float* __restrict__ h) {
    __shared__ __attribute__((aligned(16))) __bf16 At[32][36];
    __shared__ __attribute__((aligned(16))) __bf16 Bt[64][40];
    const int tid = threadIdx.x, lane = tid & 31, wave = tid >> 5;
    const int wm = wave & 1, wn = wave >> 1;
    const int p0 = blockIdx.x * 32, oc0 = blockIdx.y * 64;
    v8f acc = {};

    const int arow = tid >> 3, akl = (tid & 7) * 4;   // 4 bf16 (8B) per thread
    const int bn = tid >> 2,  bkl = (tid & 3) * 8;    // 8 bf16 (16B) per thread
    const int p = p0 + arow;
    const int oy = p / FW, ox = p - oy * FW;

    for (int tap = 0; tap < 9; ++tap) {
        const int iy = oy + tap / 3 - 1, ix = ox + tap % 3 - 1;
        const bool inb = (p < NPOS) && iy >= 0 && iy < FH && ix >= 0 && ix < FW;
        const __bf16* fr = featT + (size_t)(iy * FW + ix) * CIN;
        const __bf16* wr = Wt + ((size_t)tap * CMID + oc0 + bn) * CIN;
#pragma unroll 1
        for (int c0 = 0; c0 < CIN; c0 += 32) {
            ull av = 0;
            if (inb) av = *(const ull*)(fr + c0 + akl);
            *(ull*)&At[arow][akl] = av;
            *(uint4*)&Bt[bn][bkl] = *(const uint4*)(wr + c0 + bkl);
            __syncthreads();
            v16bf a = load_a_frag(&At[wm * 16][0], 36, lane);
            v16bf b = load_b_frag(&Bt[wn * 16][0], 40, lane);
            acc = __builtin_amdgcn_wmma_f32_16x16x32_bf16(false, a, false, b,
                                                          (short)0, acc, false, false);
            __syncthreads();
        }
    }
    const int n = oc0 + wn * 16 + (lane & 15);
    const int pbase = p0 + wm * 16 + (lane >> 4) * 8;
#pragma unroll
    for (int r = 0; r < 8; ++r) {
        int pp = pbase + r;
        if (pp < NPOS) h[n * NPOS + pp] = fmaxf(acc[r] + bias[n], 0.f);
    }
}

// ---------------------------------------------------------------------------
// Kernel 2: RPN 1x1 heads -> d_out rpn regions
// ---------------------------------------------------------------------------
__global__ void rpn_head_kernel(const float* __restrict__ h,
                                const float* __restrict__ Wl, const float* __restrict__ bl,
                                const float* __restrict__ Ws, const float* __restrict__ bsc,
                                float* __restrict__ out) {
    int t = blockIdx.x * blockDim.x + threadIdx.x;
    if (t >= NPOS * 54) return;
    int ch = t % 54, p = t / 54;
    if (ch < 36) {
        float acc = bl[ch];
        const float* w = Wl + ch * CMID;
        for (int c = 0; c < CMID; ++c) acc += h[c * NPOS + p] * w[c];
        int a = ch >> 2, coord = ch & 3;
        out[OUT_LOC + (p * 9 + a) * 4 + coord] = acc;
    } else {
        int s = ch - 36;
        float acc = bsc[s];
        const float* w = Ws + s * CMID;
        for (int c = 0; c < CMID; ++c) acc += h[c * NPOS + p] * w[c];
        int a = s >> 1, j = s & 1;
        out[OUT_SCORE + (p * 9 + a) * 2 + j] = acc;
    }
}

// ---------------------------------------------------------------------------
// Kernel 3: anchors + proposal decode + softmax fg + validity
// ---------------------------------------------------------------------------
__global__ void decode_kernel(const float* __restrict__ loc,
                              const float* __restrict__ score,
                              float* __restrict__ roi, float* __restrict__ sc) {
    int i = blockIdx.x * blockDim.x + threadIdx.x;
    if (i >= NANCH) return;
    int p = i / 9, a = i % 9;
    int y = p / FW, x = p - y * FW;
    int ratio = a / 3, scale = a % 3;
    const float scales[3] = {8.f, 16.f, 32.f};
    const float ratios[3] = {0.5f, 1.f, 2.f};
    float sr = sqrtf(ratios[ratio]);
    float ah_ = 16.f * scales[scale] * sr;
    float aw_ = 16.f * scales[scale] / sr;
    float a0 = rintf(-ah_ * 0.5f) + x * 16.f;
    float a1 = rintf(-aw_ * 0.5f) + y * 16.f;
    float a2 = rintf( ah_ * 0.5f) + x * 16.f;
    float a3 = rintf( aw_ * 0.5f) + y * 16.f;
    float aw = a2 - a0, ahh = a3 - a1;
    float cx = a0 + 0.5f * aw, cy = a1 + 0.5f * ahh;
    float dx = loc[i * 4 + 0], dy = loc[i * 4 + 1];
    float dw = loc[i * 4 + 2], dh = loc[i * 4 + 3];
    float ctx = dx * aw + cx, cty = dy * ahh + cy;
    float w = expf(dw) * aw, hh = expf(dh) * ahh;
    float x1 = fminf(fmaxf(ctx - 0.5f * w, 0.f), 800.f);
    float x2 = fminf(fmaxf(ctx + 0.5f * w, 0.f), 800.f);
    float y1 = fminf(fmaxf(cty - 0.5f * hh, 0.f), 800.f);
    float y2 = fminf(fmaxf(cty + 0.5f * hh, 0.f), 800.f);
    roi[i * 4 + 0] = x1; roi[i * 4 + 1] = y1;
    roi[i * 4 + 2] = x2; roi[i * 4 + 3] = y2;
    float s0 = score[i * 2], s1 = score[i * 2 + 1];
    float m = fmaxf(s0, s1);
    float e0 = expf(s0 - m), e1 = expf(s1 - m);
    float fg = e1 / (e0 + e1);
    bool valid = (x2 - x1 >= 16.f) && (y2 - y1 >= 16.f);
    sc[i] = valid ? fg : -__builtin_inff();
}

// ---------------------------------------------------------------------------
// top-3000 via bitonic sort on packed u64 keys
// ---------------------------------------------------------------------------
__global__ void key_init_kernel(const float* __restrict__ sc, ull* __restrict__ keys) {
    int i = blockIdx.x * blockDim.x + threadIdx.x;
    if (i >= NSORT) return;
    if (i < NANCH) {
        unsigned s = sortable_u32(sc[i]);
        keys[i] = ((ull)(~s) << 32) | (unsigned)i;
    } else {
        keys[i] = 0xFFFFFFFFFFFFFFFFull;
    }
}

__global__ void bitonic_step_kernel(ull* __restrict__ keys, int j, int k) {
    int i = blockIdx.x * blockDim.x + threadIdx.x;
    int ixj = i ^ j;
    if (ixj > i && ixj < NSORT) {
        ull a = keys[i], b = keys[ixj];
        bool up = ((i & k) == 0);
        if ((a > b) == up) { keys[i] = b; keys[ixj] = a; }
    }
}

__global__ void gather_topk_kernel(const ull* __restrict__ keys,
                                   const float* __restrict__ roi,
                                   const float* __restrict__ sc,
                                   float* __restrict__ boxes, float* __restrict__ bs) {
    int t = blockIdx.x * blockDim.x + threadIdx.x;
    if (t >= NPRE) return;
    int idx = (int)(keys[t] & 0xFFFFFFFFu);
#pragma unroll
    for (int c = 0; c < 4; ++c) boxes[t * 4 + c] = roi[idx * 4 + c];
    bs[t] = sc[idx];
}

// ---------------------------------------------------------------------------
// NMS: suppression bitmask + sequential greedy scan + stable top-300
// ---------------------------------------------------------------------------
__global__ void nms_mask_kernel(const float* __restrict__ b, unsigned* __restrict__ mask) {
    int t = blockIdx.x * blockDim.x + threadIdx.x;
    if (t >= NPRE * NMSW) return;
    int i = t / NMSW, w = t % NMSW;
    float x1 = b[i * 4], y1 = b[i * 4 + 1], x2 = b[i * 4 + 2], y2 = b[i * 4 + 3];
    float ai = (x2 - x1) * (y2 - y1);
    unsigned bits = 0;
    for (int bb = 0; bb < 32; ++bb) {
        int j = w * 32 + bb;
        if (j < NPRE && j > i) {
            float u1 = fmaxf(x1, b[j * 4]),     v1 = fmaxf(y1, b[j * 4 + 1]);
            float u2 = fminf(x2, b[j * 4 + 2]), v2 = fminf(y2, b[j * 4 + 3]);
            float inter = fmaxf(u2 - u1, 0.f) * fmaxf(v2 - v1, 0.f);
            float aj = (b[j * 4 + 2] - b[j * 4]) * (b[j * 4 + 3] - b[j * 4 + 1]);
            float iou = inter / (ai + aj - inter + 1e-9f);
            if (iou > 0.7f) bits |= 1u << bb;
        }
    }
    mask[i * NMSW + w] = bits;
}

__global__ void nms_scan_kernel(const float* __restrict__ bs,
                                const unsigned* __restrict__ mask,
                                const float* __restrict__ boxes,
                                float* __restrict__ rois) {
    __shared__ unsigned keep[NMSW];
    __shared__ int sel[NPOST];
    __shared__ unsigned char selKeep[NPOST];
    const int tid = threadIdx.x;
    for (int w = tid; w < NMSW; w += blockDim.x) {
        unsigned bits = 0;
        for (int b = 0; b < 32; ++b) {
            int i = w * 32 + b;
            if (i < NPRE && bs[i] > -1e37f) bits |= 1u << b;
        }
        keep[w] = bits;
    }
    __syncthreads();
    for (int i = 0; i < NPRE; ++i) {
        bool k = (keep[i >> 5] >> (i & 31)) & 1u;
        if (k) {
            for (int w = tid; w < NMSW; w += blockDim.x)
                keep[w] &= ~mask[i * NMSW + w];
        }
        __syncthreads();
    }
    if (tid == 0) {
        int cnt = 0;
        for (int i = 0; i < NPRE && cnt < NPOST; ++i)
            if ((keep[i >> 5] >> (i & 31)) & 1u) { sel[cnt] = i; selKeep[cnt] = 1; ++cnt; }
        for (int i = 0; i < NPRE && cnt < NPOST; ++i)
            if (!((keep[i >> 5] >> (i & 31)) & 1u)) { sel[cnt] = i; selKeep[cnt] = 0; ++cnt; }
    }
    __syncthreads();
    for (int t = tid; t < NPOST; t += blockDim.x) {
        int s = sel[t];
        bool k = selKeep[t];
#pragma unroll
        for (int c = 0; c < 4; ++c) rois[t * 4 + c] = k ? boxes[s * 4 + c] : 0.f;
    }
}

// ---------------------------------------------------------------------------
// ROI-align over raw f32 features -> bf16 FC1 A operand
// ---------------------------------------------------------------------------
__global__ void roi_align_kernel(const float* __restrict__ feat,
                                 const float* __restrict__ rois,
                                 __bf16* __restrict__ poolA) {
    int t = blockIdx.x * blockDim.x + threadIdx.x;
    if (t >= NPOST * CIN * 49) return;
    int px = t % 7, py = (t / 7) % 7, c = (t / 49) % CIN, r = t / (49 * CIN);
    float x1 = rois[r * 4 + 0] / 16.f, y1 = rois[r * 4 + 1] / 16.f;
    float x2 = rois[r * 4 + 2] / 16.f, y2 = rois[r * 4 + 3] / 16.f;
    float bw = fmaxf(x2 - x1, 1.f) / 7.f;
    float bh = fmaxf(y2 - y1, 1.f) / 7.f;
    float X = x1 + (px + 0.5f) * bw;
    float Y = y1 + (py + 0.5f) * bh;
    float x0 = floorf(X), y0 = floorf(Y);
    int x0i = min(max((int)x0, 0), FW - 1), x1i = min(x0i + 1, FW - 1);
    int y0i = min(max((int)y0, 0), FH - 1), y1i = min(y0i + 1, FH - 1);
    float wx = X - x0, wy = Y - y0;
    const float* fc = feat + c * NPOS;
    float v00 = fc[y0i * FW + x0i], v01 = fc[y0i * FW + x1i];
    float v10 = fc[y1i * FW + x0i], v11 = fc[y1i * FW + x1i];
    float o = v00 * (1 - wy) * (1 - wx) + v01 * (1 - wy) * wx +
              v10 * wy * (1 - wx) + v11 * wy * wx;
    poolA[r * POOLK + c * 49 + py * 7 + px] = f2bf(o);
}

// ---------------------------------------------------------------------------
// Skinny-M WMMA GEMM with split-K: Cpart[s] += A[300xK] @ B[KxN] over K-chunk s.
// Each block owns a 64-wide N slice and ALL 300 M rows (padded to 320):
// 8 waves = 2(M halves) x 4(N cols of 16); 10 accumulators per wave.
// -> every B element is read exactly once from HBM (B stream is the roofline
//    floor for FC1: 822 MB of f32 W_fc1 / 23.3 TB/s).
// A tile (320x32 bf16) staged by one TDM descriptor per K-step when available:
// tensor_dim1 = 300 zero-fills the 20 padding rows, pad fields give pitch 36.
// Deterministic: split-K partials reduced in fixed order by reduce_bias_kernel.
// ---------------------------------------------------------------------------
__global__ void gemm_skinny_kernel(const __bf16* __restrict__ A,
                                   const float* __restrict__ B,
                                   float* __restrict__ Cpart,
                                   int M, int N, int K) {
    __shared__ __attribute__((aligned(16))) __bf16 At[320][36];
    __shared__ __attribute__((aligned(16))) __bf16 Bt[64][36];
    const int tid = threadIdx.x, lane = tid & 31, wave = tid >> 5;
    const int wm = wave & 1, wn = wave >> 1;
    const int n0 = blockIdx.y * 64;
    const int kchunk = K / KSPLIT;
    const int kbeg = blockIdx.x * kchunk, kend = kbeg + kchunk;
    v8f acc[MT_PER_WAVE];
    v8f zero = {};
#pragma unroll
    for (int t = 0; t < MT_PER_WAVE; ++t) acc[t] = zero;
#if HAVE_TDM
    const unsigned ldsAt = (unsigned)(uintptr_t)&At[0][0];
#else
    const int arow = tid >> 3, akl = (tid & 7) * 4;
#endif

    for (int k0 = kbeg; k0 < kend; k0 += 32) {
        if (k0 + 32 < kend)  // global_prefetch_b8 on next f32 B tile
            __builtin_prefetch(&B[(size_t)(k0 + 32 + (tid >> 3)) * N + n0 + (tid & 7) * 8], 0, 1);

        // ---- A tile 320x32: TDM DMA (wave 0) or manual staging ----
#if HAVE_TDM
        if (wave == 0) {
            ull ga = (ull)(uintptr_t)A + (ull)(unsigned)k0 * 2ull;   // tile start: row 0, col k0
            u32x4 g0;
            g0[0] = 1u;                                  // count=1 (valid user D#)
            g0[1] = ldsAt;                               // lds_addr
            g0[2] = (unsigned)ga;                        // global_addr[31:0]
            g0[3] = (unsigned)(ga >> 32) | 0x80000000u;  // global_addr[56:32] | type=2
            i32x8 g1;
            g1[0] = (1 << 16)   // data_size = 2 bytes
                  | (1 << 20)   // pad_enable
                  | (3 << 22)   // pad_interval = 16 DWORDs (one 32-bf16 row)
                  | (1 << 25);  // pad_amount = 2 DWORDs -> LDS pitch 36 bf16
            g1[1] = (K & 0xFFFF) << 16;                          // tensor_dim0 lo
            g1[2] = ((K >> 16) & 0xFFFF) | ((M & 0xFFFF) << 16); // dim0 hi | dim1(=M) lo
            g1[3] = (32 << 16);                                  // dim1 hi | tile_dim0=32
            g1[4] = 320;                                         // tile_dim1=320 (rows>=M zero-filled)
            g1[5] = K;                                           // tensor_dim0_stride lo
            g1[6] = 0;
            g1[7] = 0;
            i32x4 z4 = {0, 0, 0, 0};
# if __clang_major__ >= 23
            i32x8 z8 = {0, 0, 0, 0, 0, 0, 0, 0};
            __builtin_amdgcn_tensor_load_to_lds(g0, g1, z4, z4, z8, 0);
# else
            __builtin_amdgcn_tensor_load_to_lds(g0, g1, z4, z4, 0);
# endif
            __builtin_amdgcn_s_wait_tensorcnt(0);
        }
#else
#pragma unroll
        for (int e = 0; e < 10; ++e) {
            int row = e * 32 + arow;
            ull av = 0;
            if (row < M) av = *(const ull*)(A + (size_t)row * K + k0 + akl);
            *(ull*)&At[row][akl] = av;
        }
#endif
        // ---- B tile 64n x 32k: f32 -> bf16 in flight, coalesced over n ----
#pragma unroll
        for (int e = 0; e < 8; ++e) {
            int idx = e * 256 + tid;
            int n = idx & 63, kl = idx >> 6;
            int gn = n0 + n;
            Bt[n][kl] = (gn < N) ? f2bf(B[(size_t)(k0 + kl) * N + gn]) : f2bf(0.f);
        }
        __syncthreads();
        // one B fragment reused across 10 WMMAs
        v16bf b = load_b_frag(&Bt[wn * 16][0], 36, lane);
#pragma unroll
        for (int t = 0; t < MT_PER_WAVE; ++t) {
            v16bf a = load_a_frag(&At[(wm * MT_PER_WAVE + t) * 16][0], 36, lane);
            acc[t] = __builtin_amdgcn_wmma_f32_16x16x32_bf16(false, a, false, b,
                                                             (short)0, acc[t], false, false);
        }
        __syncthreads();
    }
    const int n = n0 + wn * 16 + (lane & 15);
    if (n < N) {
        float* cp = Cpart + (size_t)blockIdx.x * M * N;
#pragma unroll
        for (int t = 0; t < MT_PER_WAVE; ++t) {
            int mb = (wm * MT_PER_WAVE + t) * 16 + (lane >> 4) * 8;
#pragma unroll
            for (int r = 0; r < 8; ++r) {
                int m = mb + r;
                if (m < M) cp[(size_t)m * N + n] = acc[t][r];
            }
        }
    }
}

// Fixed-order split-K reduction + bias + activation (bit-deterministic).
__global__ void reduce_bias_kernel(const float* __restrict__ Cpart,
                                   const float* __restrict__ bias,
                                   float* __restrict__ Cout, __bf16* __restrict__ Cbf,
                                   int M, int N, int relu) {
    int t = blockIdx.x * blockDim.x + threadIdx.x;
    if (t >= M * N) return;
    int n = t % N;
    float v = bias[n];
#pragma unroll
    for (int s = 0; s < KSPLIT; ++s) v += Cpart[(size_t)s * M * N + t];
    if (relu) v = fmaxf(v, 0.f);
    if (Cout) Cout[t] = v;
    if (Cbf)  Cbf[t] = f2bf(v);
}

// ---------------------------------------------------------------------------
// Host-side launch sequence
// ---------------------------------------------------------------------------
extern "C" void kernel_launch(void* const* d_in, const int* in_sizes, int n_in,
                              void* d_out, int out_size, void* d_ws, size_t ws_size,
                              hipStream_t stream) {
    const float* features = (const float*)d_in[0];
    const float* W_conv   = (const float*)d_in[1];
    const float* b_conv   = (const float*)d_in[2];
    const float* W_loc    = (const float*)d_in[3];
    const float* b_loc    = (const float*)d_in[4];
    const float* W_score  = (const float*)d_in[5];
    const float* b_score  = (const float*)d_in[6];
    const float* W_fc1    = (const float*)d_in[7];
    const float* b_fc1    = (const float*)d_in[8];
    const float* W_fc2    = (const float*)d_in[9];
    const float* b_fc2    = (const float*)d_in[10];
    const float* W_cls    = (const float*)d_in[11];
    const float* b_cls    = (const float*)d_in[12];
    const float* W_reg    = (const float*)d_in[13];
    const float* b_reg    = (const float*)d_in[14];
    float* out = (float*)d_out;

    char* ws = (char*)d_ws;
    size_t off = 0;
    auto carve = [&](size_t bytes) { size_t r = off; off = (off + bytes + 255) & ~(size_t)255; return r; };
    __bf16*   featT   = (__bf16*)(ws + carve((size_t)NPOS * CIN * 2));
    __bf16*   WconvT  = (__bf16*)(ws + carve((size_t)9 * CMID * CIN * 2));
    float*    h       = (float*)(ws + carve((size_t)CMID * NPOS * 4));
    float*    roi     = (float*)(ws + carve((size_t)NANCH * 4 * 4));
    float*    sc      = (float*)(ws + carve((size_t)NANCH * 4));
    ull*      keys    = (ull*)(ws + carve((size_t)NSORT * 8));
    float*    boxes3k = (float*)(ws + carve((size_t)NPRE * 4 * 4));
    float*    bs      = (float*)(ws + carve((size_t)NPRE * 4));
    unsigned* mask    = (unsigned*)(ws + carve((size_t)NPRE * NMSW * 4));
    float*    rois    = (float*)(ws + carve((size_t)NPOST * 4 * 4));
    __bf16*   poolA   = (__bf16*)(ws + carve((size_t)NPOST * POOLK * 2));
    __bf16*   h1bf    = (__bf16*)(ws + carve((size_t)NPOST * NFC * 2));
    __bf16*   h2bf    = (__bf16*)(ws + carve((size_t)NPOST * NFC * 2));
    float*    Cpart   = (float*)(ws + carve((size_t)KSPLIT * NPOST * NFC * 4));

    // 0) layout prepasses
    featT_kernel<<<(NPOS * CIN + 255) / 256, 256, 0, stream>>>(features, featT);
    wconv_reorder_kernel<<<(9 * CMID * CIN + 255) / 256, 256, 0, stream>>>(W_conv, WconvT);

    // 1) 3x3 conv + relu (WMMA implicit GEMM)
    conv3x3_wmma_kernel<<<dim3((NPOS + 31) / 32, CMID / 64), 256, 0, stream>>>(
        featT, WconvT, b_conv, h);

    // 2) RPN 1x1 heads
    rpn_head_kernel<<<(NPOS * 54 + 255) / 256, 256, 0, stream>>>(
        h, W_loc, b_loc, W_score, b_score, out);

    // 3) decode proposals
    decode_kernel<<<(NANCH + 255) / 256, 256, 0, stream>>>(
        out + OUT_LOC, out + OUT_SCORE, roi, sc);

    // 4) top-3000 (bitonic)
    key_init_kernel<<<NSORT / 256, 256, 0, stream>>>(sc, keys);
    for (int k = 2; k <= NSORT; k <<= 1)
        for (int j = k >> 1; j > 0; j >>= 1)
            bitonic_step_kernel<<<NSORT / 256, 256, 0, stream>>>(keys, j, k);
    gather_topk_kernel<<<(NPRE + 255) / 256, 256, 0, stream>>>(keys, roi, sc, boxes3k, bs);

    // 5) greedy NMS
    nms_mask_kernel<<<(NPRE * NMSW + 255) / 256, 256, 0, stream>>>(boxes3k, mask);
    nms_scan_kernel<<<1, 128, 0, stream>>>(bs, mask, boxes3k, rois);

    // 6) ROI align -> bf16 FC1 A operand
    roi_align_kernel<<<(NPOST * CIN * 49 + 255) / 256, 256, 0, stream>>>(
        features, rois, poolA);

    // 7) FC1: 300x4096, K=50176; W_fc1 streamed exactly once
    gemm_skinny_kernel<<<dim3(KSPLIT, NFC / 64), 256, 0, stream>>>(
        poolA, W_fc1, Cpart, NPOST, NFC, POOLK);
    reduce_bias_kernel<<<(NPOST * NFC + 255) / 256, 256, 0, stream>>>(
        Cpart, b_fc1, (float*)nullptr, h1bf, NPOST, NFC, 1);

    // 8) FC2: 300x4096, K=4096
    gemm_skinny_kernel<<<dim3(KSPLIT, NFC / 64), 256, 0, stream>>>(
        h1bf, W_fc2, Cpart, NPOST, NFC, NFC);
    reduce_bias_kernel<<<(NPOST * NFC + 255) / 256, 256, 0, stream>>>(
        Cpart, b_fc2, (float*)nullptr, h2bf, NPOST, NFC, 1);

    // 9) heads straight into d_out: reg (N=84) then cls (N=21)
    gemm_skinny_kernel<<<dim3(KSPLIT, 2), 256, 0, stream>>>(
        h2bf, W_reg, Cpart, NPOST, 84, NFC);
    reduce_bias_kernel<<<(NPOST * 84 + 255) / 256, 256, 0, stream>>>(
        Cpart, b_reg, out + OUT_REG, (__bf16*)nullptr, NPOST, 84, 0);
    gemm_skinny_kernel<<<dim3(KSPLIT, 1), 256, 0, stream>>>(
        h2bf, W_cls, Cpart, NPOST, 21, NFC);
    reduce_bias_kernel<<<(NPOST * 21 + 255) / 256, 256, 0, stream>>>(
        Cpart, b_cls, out + OUT_CLS, (__bf16*)nullptr, NPOST, 21, 0);

    (void)in_sizes; (void)n_in; (void)out_size; (void)ws_size;
}
